// VQEmbeddingGSSoft_44573170597891
// MI455X (gfx1250) — matrix operand order repaired and checked by reading, not compiled, not run on hardware
//
#include <hip/hip_runtime.h>

typedef __bf16 v16bf __attribute__((ext_vector_type(16)));
typedef float  v8f   __attribute__((ext_vector_type(8)));

#define NB    8          // N batches
#define MC    512        // codes per batch
#define DD    32         // embedding dim
#define HW    1024       // H*W
#define KTILE 128        // k rows per workgroup
#define LOGM  6.2383246250395075f   // ln(512)

__device__ __forceinline__ unsigned short f2bf(float f) {
    unsigned u = __float_as_uint(f);
    u += 0x7FFFu + ((u >> 16) & 1u);          // round-to-nearest-even
    return (unsigned short)(u >> 16);
}

union FragBF { v16bf v; unsigned u[8]; };

__global__ void kl_init_kernel(float* out) {
    if (threadIdx.x == 0 && blockIdx.x == 0) out[0] = 0.f;
}

__global__ __launch_bounds__(256)
void vq_gssoft_kernel(const float* __restrict__ x,
                      const float* __restrict__ emb,
                      float* __restrict__ out)
{
    // LDS: 34816 + 2048 + 9216 + 16896 + 4 = 62980 B (< 64 KB)
    __shared__ unsigned short embB[MC * 34];   // bf16 codes, row stride 34 (17 dw, gcd(17,64)=1)
    __shared__ float          eSq[MC];
    __shared__ unsigned short xT[KTILE * 36];  // bf16 x tile, row stride 36
    __shared__ float          qT[KTILE * 33];  // f32 output tile, row stride 33
    __shared__ float          klShared;

    const int tid  = threadIdx.x;
    const int n    = blockIdx.y;
    const int wgk  = blockIdx.x;               // 0..127
    const int b    = wgk >> 3;                 // 8 k-tiles per image
    const int kinb = (wgk & 7) * KTILE;        // offset within H*W

    const float* embN  = emb + (size_t)n * (MC * DD);
    const float* xbase = x   + (size_t)(b * 256 + n * 32) * HW + kinb;
    float*       qbase = out + 1 + (size_t)(b * 256 + n * 32) * HW + kinb;

    if (tid == 0) klShared = 0.f;

    // ---- stage embedding as bf16 + fp32 squared norms ----
    #pragma unroll
    for (int mm = 0; mm < 2; ++mm) {
        int m = tid * 2 + mm;
        const float* er = embN + m * DD;
        float acc = 0.f;
        #pragma unroll
        for (int d = 0; d < DD; ++d) {
            float v = er[d];
            acc += v * v;
            embB[m * 34 + d] = f2bf(v);
        }
        eSq[m] = acc;
    }

    // ---- stage x tile as bf16 (coalesced: k contiguous for fixed d) ----
    for (int i = tid; i < KTILE * DD; i += 256) {
        int d  = i >> 7;
        int kk = i & (KTILE - 1);
        xT[kk * 36 + d] = f2bf(xbase[d * HW + kk]);
    }
    __syncthreads();

    const int lane = tid & 31;
    const int wid  = tid >> 5;                 // 8 waves x 16 rows = 128 rows
    const int j    = lane & 15;
    const int hi   = lane >> 4;
    const int rowBase = wid * 16;

    // ---- A fragment: 16x32 bf16, contiguous k = 16*hi + s (reused by both passes) ----
    FragBF A;
    {
        const unsigned* xU = (const unsigned*)xT;   // 18 dwords per row
        int rb = (rowBase + j) * 18 + hi * 8;       // upper half: K += 16
        #pragma unroll
        for (int v = 0; v < 8; ++v)
            A.u[v] = xU[rb + v];                    // K sequential 0..15
    }

    const unsigned* eU = (const unsigned*)embB;     // 17 dwords per code row

    // ================= PASS 1: row max + argmax (no transcendentals) =====
    float mx[8];
    int   bi[8];
    #pragma unroll
    for (int r = 0; r < 8; ++r) { mx[r] = -__builtin_inff(); bi[r] = 0; }

    #pragma unroll 2
    for (int t = 0; t < MC / 16; ++t) {
        const int col = t * 16 + j;                 // this lane's column (N = lane&15)
        FragBF Bm;                                   // B 32x16: K sequential, upper half K += 16
        const int mb = col * 17 + hi * 8;
        #pragma unroll
        for (int v = 0; v < 8; ++v) Bm.u[v] = eU[mb + v];

        v8f c = {0.f, 0.f, 0.f, 0.f, 0.f, 0.f, 0.f, 0.f};
        c = __builtin_amdgcn_wmma_f32_16x16x32_bf16(false, A.v, false, Bm.v,
                                                    (short)0, c, false, false);
        const float es = eSq[col];
        #pragma unroll
        for (int r = 0; r < 8; ++r) {               // row M = r + 8*hi
            float lg = __builtin_fmaf(2.f, c[r], -es);
            bool gt = lg > mx[r];                   // one compare, two selects
            bi[r] = gt ? col : bi[r];               // strict > keeps first max
            mx[r] = gt ? lg  : mx[r];
        }
    }

    // cross-lane max/argmax over the 16 lanes of each half (branchless)
    #pragma unroll
    for (int msk = 1; msk < 16; msk <<= 1) {
        #pragma unroll
        for (int r = 0; r < 8; ++r) {
            float om = __shfl_xor(mx[r], msk, 32);
            int   ob = __shfl_xor(bi[r], msk, 32);
            int take = (int)(om > mx[r]) |
                       ((int)(om == mx[r]) & (int)(ob < bi[r]));  // no short-circuit
            bi[r] = take ? ob : bi[r];              // argmax: first max wins
            mx[r] = take ? om : mx[r];              // take implies om >= mx
        }
    }

    // ============ PASS 2: exp(lg - M) accumulation (M known, args <= 0) ==
    float sE[8], wE[8];
    #pragma unroll
    for (int r = 0; r < 8; ++r) { sE[r] = 0.f; wE[r] = 0.f; }

    #pragma unroll 2
    for (int t = 0; t < MC / 16; ++t) {
        const int col = t * 16 + j;
        FragBF Bm;
        const int mb = col * 17 + hi * 8;
        #pragma unroll
        for (int v = 0; v < 8; ++v) Bm.u[v] = eU[mb + v];

        v8f c = {0.f, 0.f, 0.f, 0.f, 0.f, 0.f, 0.f, 0.f};
        c = __builtin_amdgcn_wmma_f32_16x16x32_bf16(false, A.v, false, Bm.v,
                                                    (short)0, c, false, false);
        const float es = eSq[col];
        #pragma unroll
        for (int r = 0; r < 8; ++r) {
            float lg = __builtin_fmaf(2.f, c[r], -es);
            float e  = __expf(lg - mx[r]);          // <= 1, no clamp needed
            sE[r] += e;
            wE[r] = __builtin_fmaf(e, lg, wE[r]);
        }
    }

    // plain sum reduction (no exps: all partials share the same max)
    #pragma unroll
    for (int msk = 1; msk < 16; msk <<= 1) {
        #pragma unroll
        for (int r = 0; r < 8; ++r) {
            sE[r] += __shfl_xor(sE[r], msk, 32);
            wE[r] += __shfl_xor(wE[r], msk, 32);
        }
    }

    // ---- KL: sum_m p*(logp + logM) = w/s - (mx + log s) + logM per row ----
    if (j == 0) {
        float kl = 0.f;
        #pragma unroll
        for (int r = 0; r < 8; ++r)
            kl += wE[r] / sE[r] - (mx[r] + __logf(sE[r])) + LOGM;
        atomicAdd(&klShared, kl);                   // lanes 0 and 16: rows 0-7 / 8-15
    }

    // ---- gather winning fp32 embedding rows into LDS transpose tile ----
    #pragma unroll
    for (int r = 0; r < 8; ++r) {
        int row = rowBase + r + 8 * hi;
        const float* src = embN + bi[r] * DD;
        qT[row * 33 + j]      = src[j];
        qT[row * 33 + j + 16] = src[j + 16];
    }
    __syncthreads();

    // ---- coalesced store of quantized tile ----
    for (int i = tid; i < KTILE * DD; i += 256) {
        int d  = i >> 7;
        int kk = i & (KTILE - 1);
        qbase[d * HW + kk] = qT[kk * 33 + d];
    }
    if (tid == 0) atomicAdd(out, klShared * (1.f / 16.f));   // /B
}

extern "C" void kernel_launch(void* const* d_in, const int* in_sizes, int n_in,
                              void* d_out, int out_size, void* d_ws, size_t ws_size,
                              hipStream_t stream) {
    (void)in_sizes; (void)n_in; (void)out_size; (void)d_ws; (void)ws_size;
    const float* x   = (const float*)d_in[0];
    const float* emb = (const float*)d_in[1];
    float*       out = (float*)d_out;

    kl_init_kernel<<<1, 32, 0, stream>>>(out);
    dim3 grid(128, NB, 1);   // 128 k-tiles x 8 batches
    dim3 block(256, 1, 1);   // 8 wave32s
    vq_gssoft_kernel<<<grid, block, 0, stream>>>(x, emb, out);
}
